// GMixer_2894807957539
// MI455X (gfx1250) — matrix operand order repaired and checked by reading, not compiled, not run on hardware
//
#include <hip/hip_runtime.h>

// MI455X / gfx1250 — wave32, WMMA f32_16x16x32_f16 path.
//
// Problem: B=2048,T=64,E=5,S=512,D=128,EMB=8,M=8.
// The scan's carried buffer has a closed form, so all (b,t) rows are
// independent. The two chained linear layers are fused into one GEMM:
//   QKV[N=131072, 192] = statess[N,512] @ Wc[512,192] + bcomb
// with Wc = W_state^T @ [W_q;W_k;W_v]^T precomputed on device.
// statess (256 MB fp32) is streamed once with NT cache hints (read-once
// stream > 192 MB L2, so RT retention would only evict the hot 192 KB
// weight array); HBM floor ~11us @ 23.3 TB/s. The GEMM runs on
// v_wmma_f32_16x16x32_f16 (f16 inputs, f32 accumulate), which keeps it
// under the memory roofline (fp32 WMMA K=4 would be compute-bound; f16
// quantization error ~1e-4 rel through the double softmax). The tiny 8x8
// attention epilogue is fused in-kernel via LDS + wave shuffles so QKV
// (96 MB) never touches HBM.

typedef __attribute__((ext_vector_type(16))) _Float16 v16h;
typedef __attribute__((ext_vector_type(8)))  float    v8f;
typedef __attribute__((ext_vector_type(4)))  float    v4f;

#define NROWS  131072     // B*T
#define SDIM   512
#define NC     192        // q(64) | k(64) | v(64)
#define DDIM   128

// ---------------------------------------------------------------------------
// Kernel P: Wc[s][n] = sum_d W_state[d][s] * Wqkv[n][d], stored as f16 in the
// WMMA B-operand layout used here (contiguous k order, same as the A loads):
//   half index = (c*NC + n)*32 + h*16 + i   with  s = 32c + 16h + i
// (lane n in tile, lanes 0-15 hold K=0..15 of the 32-chunk, lanes 16-31 hold
//  K=16..31; v16h element i = K within the half).  Also bcomb[n] (f32).
// ---------------------------------------------------------------------------
__global__ void prep_weights(const float* __restrict__ W_state,
                             const float* __restrict__ b_state,
                             const float* __restrict__ W_q,
                             const float* __restrict__ W_k,
                             const float* __restrict__ W_v,
                             _Float16* __restrict__ HB,
                             float* __restrict__ bcomb) {
  int tid = blockIdx.x * blockDim.x + threadIdx.x;
  if (tid >= SDIM * NC) return;
  int n = tid % NC;
  int s = tid / NC;
  const float* wr = (n < 64) ? (W_q + n * DDIM)
                  : (n < 128) ? (W_k + (n - 64) * DDIM)
                              : (W_v + (n - 128) * DDIM);
  float sum = 0.f;
#pragma unroll 4
  for (int d = 0; d < DDIM; ++d) sum += W_state[d * SDIM + s] * wr[d];
  int c = s >> 5, hh = (s >> 4) & 1, i = s & 15;
  HB[(c * NC + n) * 32 + hh * 16 + i] = (_Float16)sum;
  if (s == 0) {  // tid < 192: one thread per n does the bias fold
    float bs = 0.f;
#pragma unroll 4
    for (int d = 0; d < DDIM; ++d) bs += b_state[d] * wr[d];
    bcomb[n] = bs;
  }
}

// -------------------------- epilogue helpers -------------------------------
__device__ __forceinline__ float softmax8(float x) {
  float mx = x;
  mx = fmaxf(mx, __shfl_xor(mx, 1, 8));
  mx = fmaxf(mx, __shfl_xor(mx, 2, 8));
  mx = fmaxf(mx, __shfl_xor(mx, 4, 8));
  float ex = __expf(x - mx);
  float sm = ex;
  sm += __shfl_xor(sm, 1, 8);
  sm += __shfl_xor(sm, 2, 8);
  sm += __shfl_xor(sm, 4, 8);
  return ex / sm;
}

// butterfly 8x8 transpose across an 8-lane group:
// in : X[r] on lane l == M[l][r];  out: X[r] on lane l == M[r][l]
__device__ __forceinline__ void transpose8(float X[8], int f) {
#pragma unroll
  for (int s = 0; s < 3; ++s) {
    const int d = 1 << s;
    const bool sel = (f >> s) & 1;
#pragma unroll
    for (int r0 = 0; r0 < 8; ++r0) {
      if ((r0 >> s) & 1) continue;
      const int r1 = r0 + d;
      float lo = X[r0], hi = X[r1];
      float send = sel ? lo : hi;
      float recv = __shfl_xor(send, d, 8);
      X[r0] = sel ? recv : lo;
      X[r1] = sel ? hi : recv;
    }
  }
}

// ---------------------------------------------------------------------------
// Kernel G: 1 wave = 16 rows x all 192 cols (12 WMMA accumulators).
// Then fused per-row 8x8 attention (8 lanes per row, 4 rows per wave pass).
// ---------------------------------------------------------------------------
__global__ __launch_bounds__(128) void fused_gemm_attn(
    const float* __restrict__ aq,     // agent_qss [B,T,5]
    const float* __restrict__ st,     // statess   [B,T,512]
    const float* __restrict__ pos,    // [1,8,8]
    const _Float16* __restrict__ HB,  // packed Wc (B-operand layout)
    const float* __restrict__ bcomb,  // [192]
    const float* __restrict__ Wqm,    // W_qmap [1,8]
    const float* __restrict__ bqm,    // b_qmap [1]
    float* __restrict__ dout) {       // [B,T]
  __shared__ float lds[4][NC * 16];   // per-wave QKV tile, col-major [n][row]

  const int lane = threadIdx.x & 31;
  const int w    = threadIdx.x >> 5;
  const int tile = blockIdx.x * 4 + w;
  const int rowbase = tile * 16;
  const int lr = lane & 15;
  const int hh = lane >> 4;

  // ---------------- GEMM: QKV = states @ Wc ----------------
  // A operand (16x32 f16): lane(lr,hh) holds K = 32c + 16*hh + i in
  // elements i = 0..15.
  const float* aBase = st + (size_t)(rowbase + lr) * SDIM + hh * 16;

  v8f acc[12];
  {
    v8f z = {0.f, 0.f, 0.f, 0.f, 0.f, 0.f, 0.f, 0.f};
#pragma unroll
    for (int j = 0; j < 12; ++j) acc[j] = z;
  }

  for (int c = 0; c < 16; ++c) {
    // statess is a 256 MB read-once stream: non-temporal loads keep the hot
    // 192 KB weight array resident in WGP$/L2 instead of being evicted.
    const v4f* ap  = (const v4f*)(aBase + 32 * c);
    const v4f* ap2 = (const v4f*)(aBase + 32 * c + 8);
    v4f x0 = __builtin_nontemporal_load(ap);
    v4f x1 = __builtin_nontemporal_load(ap + 1);
    v4f x2 = __builtin_nontemporal_load(ap2);
    v4f x3 = __builtin_nontemporal_load(ap2 + 1);
    v16h a;
    a[0]  = (_Float16)x0[0]; a[1]  = (_Float16)x0[1];
    a[2]  = (_Float16)x0[2]; a[3]  = (_Float16)x0[3];
    a[4]  = (_Float16)x1[0]; a[5]  = (_Float16)x1[1];
    a[6]  = (_Float16)x1[2]; a[7]  = (_Float16)x1[3];
    a[8]  = (_Float16)x2[0]; a[9]  = (_Float16)x2[1];
    a[10] = (_Float16)x2[2]; a[11] = (_Float16)x2[3];
    a[12] = (_Float16)x3[0]; a[13] = (_Float16)x3[1];
    a[14] = (_Float16)x3[2]; a[15] = (_Float16)x3[3];
#pragma unroll
    for (int j = 0; j < 12; ++j) {
      v16h b = *(const v16h*)(HB + ((size_t)(c * NC + 16 * j + lr) * 32 + hh * 16));
      acc[j] = __builtin_amdgcn_wmma_f32_16x16x32_f16(
          false, a, false, b, (short)0, acc[j], false, false);
    }
  }

  // bias add + spill tile to LDS, col-major: lds[n*16 + localRow].
  // C layout: vgpr r, lanes 0-15 -> row r, lanes 16-31 -> row r+8.
  float* Lq = &lds[w][0];
#pragma unroll
  for (int j = 0; j < 12; ++j) {
    float bb = bcomb[16 * j + lr];
    float* dst = Lq + (16 * j + lr) * 16 + hh * 8;
    v4f lo = {acc[j][0] + bb, acc[j][1] + bb, acc[j][2] + bb, acc[j][3] + bb};
    v4f hi = {acc[j][4] + bb, acc[j][5] + bb, acc[j][6] + bb, acc[j][7] + bb};
    *(v4f*)dst       = lo;
    *(v4f*)(dst + 4) = hi;
  }
  __syncthreads();

  // ---------------- fused attention epilogue ----------------
  const int f   = lane & 7;   // matrix column / lane within group
  const int grp = lane >> 3;  // 4 rows handled per pass
  const float invs = 0.66874030497642202f;  // 5^(-1/4): K,Q each / e^0.25

  float wq[8];
#pragma unroll
  for (int m = 0; m < 8; ++m) wq[m] = Wqm[m];
  const float bq = bqm[0];

  float C[8];  // prefix sums of pos rows, column f
  {
    float p = 0.f;
#pragma unroll
    for (int m = 0; m < 8; ++m) { p += pos[m * 8 + f]; C[m] = p; }
  }

  for (int pass = 0; pass < 4; ++pass) {
    const int rl  = pass * 4 + grp;
    const int row = rowbase + rl;
    const int bb  = row >> 6;
    const int t   = row & 63;

    float qc[8], kc[8], vc[8];  // columns of the 8x8 q/k/v matrices
#pragma unroll
    for (int e = 0; e < 8; ++e) {
      qc[e] = Lq[(e * 8 + f) * 16 + rl] * invs;
      kc[e] = Lq[(64 + e * 8 + f) * 16 + rl] * invs;
      vc[e] = Lq[(128 + e * 8 + f) * 16 + rl];
    }

    // closed-form recurrent buffer, column f:
    // buf[m] = (m<=t ? pad(aq[t-m]) : 0) + C[m] - (m>t ? C[m-t-1] : 0)
    float bufc[8];
#pragma unroll
    for (int m = 0; m < 8; ++m) {
      float v = C[m] - ((m > t) ? C[m - t - 1] : 0.f);
      if (m <= t && f < 5) v += aq[((size_t)(bb << 6) + (t - m)) * 5 + f];
      bufc[m] = v;
    }

    // K/Q/V[m][f] = sum_e buf[m][e] * {k,q,v}[e][f]  (buf[m][e] from lane e)
    float K[8], Q[8], V[8];
#pragma unroll
    for (int m = 0; m < 8; ++m) {
      float kk = 0.f, qq = 0.f, vv = 0.f;
#pragma unroll
      for (int e = 0; e < 8; ++e) {
        float bv = __shfl(bufc[m], e, 8);
        kk += bv * kc[e];
        qq += bv * qc[e];
        vv += bv * vc[e];
      }
      K[m] = kk; Q[m] = qq; V[m] = vv;
    }

    transpose8(K, f);  // K[r] on lane n == K[n][r]
    transpose8(Q, f);  // Q[r] on lane m == Q[m][r]

    // dot[m][n] = sum_r Q[m][r]*K[n][r], lane = n; then softmax over n
    float dotc[8];
#pragma unroll
    for (int m = 0; m < 8; ++m) {
      float s2 = 0.f;
#pragma unroll
      for (int r = 0; r < 8; ++r) s2 += __shfl(Q[r], m, 8) * K[r];
      dotc[m] = s2;
    }
#pragma unroll
    for (int m = 0; m < 8; ++m) dotc[m] = softmax8(dotc[m]);

    // out[m][f] = sum_n dot[m][n]*V[n][f]; then softmax over f
    float outc[8];
#pragma unroll
    for (int m = 0; m < 8; ++m) {
      float s2 = 0.f;
#pragma unroll
      for (int n = 0; n < 8; ++n) s2 += __shfl(dotc[m], n, 8) * V[n];
      outc[m] = s2;
    }
#pragma unroll
    for (int m = 0; m < 8; ++m) outc[m] = softmax8(outc[m]);

    // q_tot = sum_m out[m][0]*W_qmap[m] + b_qmap ; time order is reversed
    float qt = bq;
#pragma unroll
    for (int m = 0; m < 8; ++m) qt += outc[m] * wq[m];
    if (f == 0) __builtin_nontemporal_store(qt, &dout[(bb << 6) + (63 - t)]);
  }
}

// ---------------------------------------------------------------------------
extern "C" void kernel_launch(void* const* d_in, const int* in_sizes, int n_in,
                              void* d_out, int out_size, void* d_ws, size_t ws_size,
                              hipStream_t stream) {
  (void)in_sizes; (void)n_in; (void)out_size; (void)ws_size;
  const float* aq      = (const float*)d_in[0];  // agent_qss [2048,64,5]
  const float* st      = (const float*)d_in[1];  // statess   [2048,64,512]
  const float* pos     = (const float*)d_in[2];  // [1,8,8]
  const float* W_state = (const float*)d_in[3];  // [128,512]
  const float* b_state = (const float*)d_in[4];  // [128]
  const float* W_q     = (const float*)d_in[5];  // [64,128]
  const float* W_k     = (const float*)d_in[6];  // [64,128]
  const float* W_v     = (const float*)d_in[7];  // [64,128]
  const float* W_qmap  = (const float*)d_in[8];  // [1,8]
  const float* b_qmap  = (const float*)d_in[9];  // [1]
  float* out = (float*)d_out;

  _Float16* HB  = (_Float16*)d_ws;                           // 192 KB
  float* bcomb  = (float*)((char*)d_ws + (size_t)SDIM * NC * sizeof(_Float16));

  hipLaunchKernelGGL(prep_weights, dim3(384), dim3(256), 0, stream,
                     W_state, b_state, W_q, W_k, W_v, HB, bcomb);

  // 131072 rows / 16 rows-per-wave = 8192 waves; 4 waves per block.
  hipLaunchKernelGGL(fused_gemm_attn, dim3(2048), dim3(128), 0, stream,
                     aq, st, pos, HB, bcomb, W_qmap, b_qmap, out);
}
